// RelationNetwork_47304769798267
// MI455X (gfx1250) — matrix operand order repaired and checked by reading, not compiled, not run on hardware
//
#include <hip/hip_runtime.h>
#include <hip/hip_bf16.h>
#include <cstdint>
#include <cstddef>

typedef _Float16 half_t;
typedef __attribute__((ext_vector_type(16))) _Float16 v16h;
typedef __attribute__((ext_vector_type(8)))  float    v8f;

#define BN_EPS 1e-5f

// problem sizes (fixed by reference)
#define BATCH 64
#define NOBJ  64
#define DIM   64
#define NPAIR (BATCH * NOBJ * NOBJ)   // 262144
#define TSZ   128                     // 2*DIM
#define HGD   256
#define OUTD  128

// ---------------- workspace layout (bytes) ----------------
#define OFF_A    0ull                 // bufA: 262144*256*4 = 256 MB (preact L0, reused for L2 out)
#define OFF_B    268435456ull         // bufB: 256 MB (preact L1)
#define OFF_XH   536870912ull         // x in f16: 512 KB
#define OFF_WT0  537395200ull         // Wt0 [256][128] f16 = 64 KB
#define OFF_WT1  537460736ull         // Wt1 [256][256] f16 = 128 KB
#define OFF_WT2  537591808ull         // Wt2 [128][256] f16 = 64 KB
#define OFF_PSUM 537657344ull         // partial sums: 512*256 f32 = 512 KB
#define OFF_PSQ  538181632ull         // partial sumsq: 512 KB
#define OFF_SC   538705920ull         // scale/shift: 1280 f32
#define OFF_SBUF 538711040ull         // s buffer 64*128 f32 = 32 KB
#define OFF_PS2  538743808ull         // pair-sum partials 64*16*128 f32 = 512 KB
#define NPART    512
#define PCHUNK   16                   // pair-sum chunks (4096/PCHUNK = 256 rows each)

// ---------------- small helper kernels ----------------
__global__ void f32_to_f16_kernel(const float* __restrict__ in, half_t* __restrict__ out, int n) {
  int i = blockIdx.x * blockDim.x + threadIdx.x;
  if (i < n) out[i] = (half_t)in[i];
}

// Wt[n][k] = (f16) W[k][n]   (W is K x N row-major)
__global__ void w_transpose_f16_kernel(const float* __restrict__ W, half_t* __restrict__ Wt, int K, int N) {
  int n = blockIdx.x;
  int k = threadIdx.x;
  if (k < K) Wt[(size_t)n * K + k] = (half_t)W[(size_t)k * N + n];
}

// deterministic 2-stage per-channel stats: stage 1 (per-block partials)
__global__ void stats_reduce_kernel(const float* __restrict__ H, int M, int C,
                                    float* __restrict__ psum, float* __restrict__ psq) {
  int c = threadIdx.x;                 // blockDim.x == C
  int rows = (M + gridDim.x - 1) / gridDim.x;
  int r0 = blockIdx.x * rows;
  int r1 = r0 + rows; if (r1 > M) r1 = M;
  float s = 0.f, q = 0.f;
  for (int r = r0; r < r1; ++r) {
    float v = H[(size_t)r * C + c];
    s += v; q += v * v;
  }
  psum[(size_t)blockIdx.x * C + c] = s;
  psq [(size_t)blockIdx.x * C + c] = q;
}

// stage 2: fold partials, produce BN scale/shift
__global__ void bn_finalize_kernel(int C, float invM,
                                   const float* __restrict__ psum, const float* __restrict__ psq,
                                   const float* __restrict__ gamma, const float* __restrict__ beta,
                                   float* __restrict__ scale, float* __restrict__ shift) {
  int c = threadIdx.x;
  if (c >= C) return;
  float s = 0.f, q = 0.f;
  for (int p = 0; p < NPART; ++p) {
    s += psum[(size_t)p * C + c];
    q += psq [(size_t)p * C + c];
  }
  float mu  = s * invM;
  float var = q * invM - mu * mu;
  float sc  = gamma[c] * rsqrtf(var + BN_EPS);
  scale[c] = sc;
  shift[c] = beta[c] - mu * sc;
}

// ---------------- async staging helpers ----------------
__device__ __forceinline__ void async_copy32B(unsigned lds_addr, const void* gsrc) {
  unsigned long long ga = (unsigned long long)gsrc;
  // GLOBAL_LOAD_ASYNC_TO_LDS_B128: async global -> LDS copy (16B per lane)
  asm volatile("global_load_async_to_lds_b128 %0, %1, off" :: "v"(lds_addr), "v"(ga) : "memory");
  asm volatile("global_load_async_to_lds_b128 %0, %1, off offset:16" :: "v"(lds_addr), "v"(ga) : "memory");
}
__device__ __forceinline__ void wait_async0() {
  asm volatile("s_wait_asynccnt 0x0" ::: "memory");
}

// ---------------- fused GEMM (+pairing / +BN-ReLU on A) ----------------
// MODE 0: A rows built from x (f16) as concat(x[b,i,:], x[b,j,:])  (async DMA staged)
// MODE 1: A rows = relu(Hprev*scale+shift) computed on the fly from f32 preact
// Wt is [N][K] f16 (pre-transposed weights; async DMA staged). Out = A@W + bias (f32).
template <int MODE>
__device__ __forceinline__ void stage_tiles(half_t (*At)[40], half_t (*Bt)[40],
                                            const void* Asrc, const half_t* Wt,
                                            const float* scale, const float* shift,
                                            int gr, int gn, int srow, int seg, int kk, int K,
                                            int pb, int pi, int pj) {
  // ---- A tile (128 x 32 f16) ----
  if (MODE == 0) {
    const half_t* xh = (const half_t*)Asrc;
    const half_t* src = (kk < DIM)
        ? (xh + ((size_t)(pb * NOBJ + pi)) * DIM + kk)
        : (xh + ((size_t)(pb * NOBJ + pj)) * DIM + (kk - DIM));
    async_copy32B((unsigned)(uintptr_t)&At[srow][seg * 16], src);
  } else {
    const float* Hp  = (const float*)Asrc;
    const float* src = Hp + (size_t)gr * K + kk;
    union { half_t h[16]; uint4 u[2]; } pk;
#pragma unroll
    for (int h4 = 0; h4 < 16; h4 += 4) {
      float4 hv = *(const float4*)(src + h4);
      float vv[4] = {hv.x, hv.y, hv.z, hv.w};
#pragma unroll
      for (int e = 0; e < 4; ++e) {
        int kc = kk + h4 + e;
        float v = vv[e] * scale[kc] + shift[kc];
        pk.h[h4 + e] = (half_t)fmaxf(v, 0.f);
      }
    }
    uint4* dst = (uint4*)&At[srow][seg * 16];
    dst[0] = pk.u[0];
    dst[1] = pk.u[1];
  }
  // ---- B tile (128 cols x 32 k, from Wt[N][K]) ----
  async_copy32B((unsigned)(uintptr_t)&Bt[srow][seg * 16], Wt + (size_t)gn * K + kk);
}

template <int MODE>
__global__ void __launch_bounds__(256)
gemm_bn_kernel(const void* __restrict__ Asrc, const half_t* __restrict__ Wt,
               const float* __restrict__ bias, const float* __restrict__ scale,
               const float* __restrict__ shift, float* __restrict__ Out,
               int M, int K, int N) {
  __shared__ alignas(16) half_t Atile[2][128][40];   // [buf][m][k], pitch 40 halves (80 B)
  __shared__ alignas(16) half_t Btile[2][128][40];   // [buf][n][k]

  const int t      = threadIdx.x;
  const int wave   = t >> 5;
  const int lane   = t & 31;
  const int lane16 = lane & 15;
  const int lhalf  = lane >> 4;
  const int wm     = wave & 3;   // 32-row strip within 128-row block tile
  const int wn     = wave >> 2;  // 64-col strip within 128-col block tile

  const int srow = t >> 1;       // staging row 0..127
  const int seg  = t & 1;        // which 16-half segment of the 32-wide k tile
  const int gr   = blockIdx.y * 128 + srow;   // global A row for staging
  const int gn   = blockIdx.x * 128 + srow;   // global B col for staging

  const v8f vzero = {0.f, 0.f, 0.f, 0.f, 0.f, 0.f, 0.f, 0.f};
  v8f acc[2][4];
#pragma unroll
  for (int mi = 0; mi < 2; ++mi)
#pragma unroll
    for (int ni = 0; ni < 4; ++ni) acc[mi][ni] = vzero;

  int pb = 0, pi = 0, pj = 0;
  if (MODE == 0) { pb = gr >> 12; pi = (gr >> 6) & 63; pj = gr & 63; }

  // prologue: stage first K tile into buffer 0
  stage_tiles<MODE>(Atile[0], Btile[0], Asrc, Wt, scale, shift,
                    gr, gn, srow, seg, seg * 16, K, pb, pi, pj);
  wait_async0();
  __syncthreads();

  int cur = 0;
  for (int kt = 0; kt < K; kt += 32) {
    const bool more = (kt + 32) < K;
    if (more) {
      stage_tiles<MODE>(Atile[cur ^ 1], Btile[cur ^ 1], Asrc, Wt, scale, shift,
                        gr, gn, srow, seg, (kt + 32) + seg * 16, K, pb, pi, pj);
    }

    // ---- fragments per documented wave32 16-bit layouts ----
    // A 16x32: lanes0-15 hold K {0..7,16..23}; lanes16-31 hold K {8..15,24..31}
    v16h af[2];
#pragma unroll
    for (int mi = 0; mi < 2; ++mi) {
      int mr = wm * 32 + mi * 16 + lane16;
      union { uint4 u[2]; v16h v; } ua;
      ua.u[0] = *(const uint4*)&Atile[cur][mr][8 * lhalf];
      ua.u[1] = *(const uint4*)&Atile[cur][mr][16 + 8 * lhalf];
      af[mi] = ua.v;
    }
    // B 32x16: lanes0-15 hold K {0..7,16..23}; lanes16-31 hold K {8..15,24..31} (Btile is [n][k])
    v16h bf[4];
#pragma unroll
    for (int ni = 0; ni < 4; ++ni) {
      int nc = wn * 64 + ni * 16 + lane16;
      union { uint4 u[2]; v16h v; } ub;
      ub.u[0] = *(const uint4*)&Btile[cur][nc][8 * lhalf];
      ub.u[1] = *(const uint4*)&Btile[cur][nc][16 + 8 * lhalf];
      bf[ni] = ub.v;
    }

#pragma unroll
    for (int mi = 0; mi < 2; ++mi)
#pragma unroll
      for (int ni = 0; ni < 4; ++ni)
        acc[mi][ni] = __builtin_amdgcn_wmma_f32_16x16x32_f16(
            false, af[mi], false, bf[ni], (short)0, acc[mi][ni], false, false);

    if (more) {
      wait_async0();       // own async prefetch into buf^1 done
      __syncthreads();     // everyone's staging done; reads of cur finished above
      cur ^= 1;
    }
  }

  // ---- epilogue: + bias, store f32 preactivations ----
  // C/D layout: VGPR v -> M = v + 8*(lane>>4); N = lane&15
#pragma unroll
  for (int mi = 0; mi < 2; ++mi) {
    int rbase = blockIdx.y * 128 + wm * 32 + mi * 16 + 8 * lhalf;
#pragma unroll
    for (int ni = 0; ni < 4; ++ni) {
      int ncol = blockIdx.x * 128 + wn * 64 + ni * 16 + lane16;
      float bv = bias[ncol];
#pragma unroll
      for (int v = 0; v < 8; ++v)
        Out[(size_t)(rbase + v) * N + ncol] = acc[mi][ni][v] + bv;
    }
  }
}

// ---------------- BN2+ReLU fused pair-sum (partials, deterministic) ----------------
__global__ void pair_sum_part_kernel(const float* __restrict__ H2, const float* __restrict__ scale,
                                     const float* __restrict__ shift, float* __restrict__ part) {
  int b  = blockIdx.x;       // 0..63
  int ch = blockIdx.y;       // 0..PCHUNK-1
  int c  = threadIdx.x;      // 0..127
  const int rows = (NOBJ * NOBJ) / PCHUNK;   // 256
  float sc = scale[c], sh = shift[c];
  const float* base = H2 + ((size_t)b * (NOBJ * NOBJ) + (size_t)ch * rows) * TSZ + c;
  float acc = 0.f;
  for (int p = 0; p < rows; ++p) {
    float v = base[(size_t)p * TSZ] * sc + sh;
    acc += fmaxf(v, 0.f);
  }
  part[((size_t)b * PCHUNK + ch) * TSZ + c] = acc;
}

__global__ void pair_sum_final_kernel(const float* __restrict__ part, float* __restrict__ s) {
  int b = blockIdx.x;        // 0..63
  int c = threadIdx.x;       // 0..127
  float acc = 0.f;
  for (int ch = 0; ch < PCHUNK; ++ch)
    acc += part[((size_t)b * PCHUNK + ch) * TSZ + c];
  s[b * TSZ + c] = acc;
}

// ---------------- F head: two Linear+BN(64 rows)+ReLU, then final Linear ----------------
__global__ void fhead_kernel(const float* __restrict__ s,
                             const float* __restrict__ fW0, const float* __restrict__ fb0,
                             const float* __restrict__ fg0, const float* __restrict__ fB0,
                             const float* __restrict__ fW1, const float* __restrict__ fb1,
                             const float* __restrict__ fg1, const float* __restrict__ fB1,
                             const float* __restrict__ foW, const float* __restrict__ fob,
                             float* __restrict__ out) {
  __shared__ float S[BATCH][TSZ];
  __shared__ float H[BATCH][TSZ];
  int c = threadIdx.x;  // 0..127
  for (int r = 0; r < BATCH; ++r) S[r][c] = s[r * TSZ + c];
  __syncthreads();

  for (int l = 0; l < 2; ++l) {
    const float* W  = l ? fW1 : fW0;
    const float* bb = l ? fb1 : fb0;
    const float* g  = l ? fg1 : fg0;
    const float* be = l ? fB1 : fB0;
    float sum = 0.f, sq = 0.f;
    for (int r = 0; r < BATCH; ++r) {
      float a = bb[c];
      for (int k = 0; k < TSZ; ++k) a += S[r][k] * W[k * TSZ + c];
      H[r][c] = a;
      sum += a; sq += a * a;
    }
    float mu  = sum * (1.f / BATCH);
    float var = sq * (1.f / BATCH) - mu * mu;
    float sc  = g[c] * rsqrtf(var + BN_EPS);
    float sh  = be[c] - mu * sc;
    __syncthreads();   // everyone done reading S
    for (int r = 0; r < BATCH; ++r) S[r][c] = fmaxf(H[r][c] * sc + sh, 0.f);
    __syncthreads();   // S ready for next layer
  }
  for (int r = 0; r < BATCH; ++r) {
    float o = fob[c];
    for (int k = 0; k < TSZ; ++k) o += S[r][k] * foW[k * OUTD + c];
    out[r * OUTD + c] = o;
  }
}

// ---------------- launch ----------------
extern "C" void kernel_launch(void* const* d_in, const int* in_sizes, int n_in,
                              void* d_out, int out_size, void* d_ws, size_t ws_size,
                              hipStream_t stream) {
  const float* x   = (const float*)d_in[0];
  const float* gW0 = (const float*)d_in[1];
  const float* gb0 = (const float*)d_in[2];
  const float* gg0 = (const float*)d_in[3];
  const float* gB0 = (const float*)d_in[4];
  const float* gW1 = (const float*)d_in[5];
  const float* gb1 = (const float*)d_in[6];
  const float* gg1 = (const float*)d_in[7];
  const float* gB1 = (const float*)d_in[8];
  const float* gW2 = (const float*)d_in[9];
  const float* gb2 = (const float*)d_in[10];
  const float* gg2 = (const float*)d_in[11];
  const float* gB2 = (const float*)d_in[12];
  const float* fW0 = (const float*)d_in[13];
  const float* fb0 = (const float*)d_in[14];
  const float* fg0 = (const float*)d_in[15];
  const float* fB0 = (const float*)d_in[16];
  const float* fW1 = (const float*)d_in[17];
  const float* fb1 = (const float*)d_in[18];
  const float* fg1 = (const float*)d_in[19];
  const float* fB1 = (const float*)d_in[20];
  const float* foW = (const float*)d_in[21];
  const float* fob = (const float*)d_in[22];

  char* ws = (char*)d_ws;
  float*  bufA = (float*)(ws + OFF_A);
  float*  bufB = (float*)(ws + OFF_B);
  half_t* xh   = (half_t*)(ws + OFF_XH);
  half_t* Wt0  = (half_t*)(ws + OFF_WT0);
  half_t* Wt1  = (half_t*)(ws + OFF_WT1);
  half_t* Wt2  = (half_t*)(ws + OFF_WT2);
  float*  psum = (float*)(ws + OFF_PSUM);
  float*  psq  = (float*)(ws + OFF_PSQ);
  float*  scsh = (float*)(ws + OFF_SC);
  float*  sc0 = scsh + 0,    *sh0 = scsh + 256;
  float*  sc1 = scsh + 512,  *sh1 = scsh + 768;
  float*  sc2 = scsh + 1024, *sh2 = scsh + 1152;
  float*  sbuf = (float*)(ws + OFF_SBUF);
  float*  ppart = (float*)(ws + OFF_PS2);
  float*  outp = (float*)d_out;

  const float invM = 1.0f / (float)NPAIR;

  // precision conversion
  f32_to_f16_kernel<<<dim3((NPAIR + 255) / 256), dim3(256), 0, stream>>>(x, xh, NPAIR);
  w_transpose_f16_kernel<<<dim3(HGD), dim3(256), 0, stream>>>(gW0, Wt0, TSZ, HGD);
  w_transpose_f16_kernel<<<dim3(HGD), dim3(256), 0, stream>>>(gW1, Wt1, HGD, HGD);
  w_transpose_f16_kernel<<<dim3(TSZ), dim3(256), 0, stream>>>(gW2, Wt2, HGD, TSZ);

  // G layer 0: pairs(x) @ W0 + b0 -> bufA (262144 x 256)
  gemm_bn_kernel<0><<<dim3(HGD / 128, NPAIR / 128), dim3(256), 0, stream>>>(
      (const void*)xh, Wt0, gb0, nullptr, nullptr, bufA, NPAIR, TSZ, HGD);
  stats_reduce_kernel<<<dim3(NPART), dim3(HGD), 0, stream>>>(bufA, NPAIR, HGD, psum, psq);
  bn_finalize_kernel<<<dim3(1), dim3(HGD), 0, stream>>>(HGD, invM, psum, psq, gg0, gB0, sc0, sh0);

  // G layer 1: relu(bn(bufA)) @ W1 + b1 -> bufB (262144 x 256)
  gemm_bn_kernel<1><<<dim3(HGD / 128, NPAIR / 128), dim3(256), 0, stream>>>(
      (const void*)bufA, Wt1, gb1, sc0, sh0, bufB, NPAIR, HGD, HGD);
  stats_reduce_kernel<<<dim3(NPART), dim3(HGD), 0, stream>>>(bufB, NPAIR, HGD, psum, psq);
  bn_finalize_kernel<<<dim3(1), dim3(HGD), 0, stream>>>(HGD, invM, psum, psq, gg1, gB1, sc1, sh1);

  // G layer 2: relu(bn(bufB)) @ W2 + b2 -> bufA (262144 x 128)
  gemm_bn_kernel<1><<<dim3(TSZ / 128, NPAIR / 128), dim3(256), 0, stream>>>(
      (const void*)bufB, Wt2, gb2, sc1, sh1, bufA, NPAIR, HGD, TSZ);
  stats_reduce_kernel<<<dim3(NPART), dim3(TSZ), 0, stream>>>(bufA, NPAIR, TSZ, psum, psq);
  bn_finalize_kernel<<<dim3(1), dim3(TSZ), 0, stream>>>(TSZ, invM, psum, psq, gg2, gB2, sc2, sh2);

  // sum over pairs with BN2+ReLU fused (deterministic 2-stage)
  pair_sum_part_kernel<<<dim3(BATCH, PCHUNK), dim3(TSZ), 0, stream>>>(bufA, sc2, sh2, ppart);
  pair_sum_final_kernel<<<dim3(BATCH), dim3(TSZ), 0, stream>>>(ppart, sbuf);

  // F head (tiny)
  fhead_kernel<<<dim3(1), dim3(TSZ), 0, stream>>>(sbuf, fW0, fb0, fg0, fB0,
                                                  fW1, fb1, fg1, fB1, foW, fob, outp);
  (void)in_sizes; (void)n_in; (void)out_size; (void)ws_size;
}
